// SA_Fuse_31808527794657
// MI455X (gfx1250) — matrix operand order repaired and checked by reading, not compiled, not run on hardware
//
#include <hip/hip_runtime.h>

typedef __attribute__((ext_vector_type(16))) _Float16 v16h;
typedef __attribute__((ext_vector_type(8)))  _Float16 v8h;
typedef __attribute__((ext_vector_type(8)))  float    v8f;

#define Bq 4
#define Cc 256
#define Nn 4096
#define Dq 32

// DPP16 lane exchange (stays within 16-lane rows => halves of the wave are
// independent, matching the WMMA D-layout where rows m=r+8*(lane/16)).
// 0xB1 = quad_perm(1,0,3,2) ~ xor1 ; 0x4E = quad_perm(2,3,0,1) ~ xor2
// 0x141 = row_half_mirror   ~ xor7 ; 0x140 = row_mirror        ~ xor15
template <int CTRL>
__device__ __forceinline__ float dpp_mov(float x) {
  int xi = __builtin_bit_cast(int, x);
  int r  = __builtin_amdgcn_update_dpp(xi, xi, CTRL, 0xF, 0xF, true);
  return __builtin_bit_cast(float, r);
}

__device__ __forceinline__ float row_max16(float x) {
  x = fmaxf(x, dpp_mov<0xB1>(x));
  x = fmaxf(x, dpp_mov<0x4E>(x));
  x = fmaxf(x, dpp_mov<0x141>(x));
  x = fmaxf(x, dpp_mov<0x140>(x));
  return x;
}
__device__ __forceinline__ float row_sum16(float x) {
  x += dpp_mov<0xB1>(x);
  x += dpp_mov<0x4E>(x);
  x += dpp_mov<0x141>(x);
  x += dpp_mov<0x140>(x);
  return x;
}

// ---------------------------------------------------------------------------
// Stage 1a: q/k projections -> f16, stored position-major [B][N][32]
// ---------------------------------------------------------------------------
__global__ __launch_bounds__(256) void qk_proj_kernel(
    const float* __restrict__ ef, const float* __restrict__ sf,
    const float* __restrict__ At,
    const float* __restrict__ Qw, const float* __restrict__ Qb,
    const float* __restrict__ Kw, const float* __restrict__ Kb,
    _Float16* __restrict__ Qh, _Float16* __restrict__ Kh)
{
  int tid = blockIdx.x * 256 + threadIdx.x;   // one thread per (b, n)
  int b = tid >> 12;
  int n = tid & (Nn - 1);

  float qa[Dq], ka[Dq];
#pragma unroll
  for (int d = 0; d < Dq; ++d) { qa[d] = Qb[d]; ka[d] = Kb[d]; }

  for (int c = 0; c < Cc; ++c) {
    float e = ef[((size_t)b * Cc + c) * Nn + n];
    float s = sf[((size_t)b * Cc + c) * Nn + n];
    float a = At[((size_t)b * Cc + c) * Nn + n];
#pragma unroll
    for (int d = 0; d < Dq; ++d) {
      qa[d] = fmaf(Qw[d * 512 + c], e, qa[d]);
      qa[d] = fmaf(Qw[d * 512 + 256 + c], s, qa[d]);
      ka[d] = fmaf(Kw[d * 256 + c], a, ka[d]);
    }
  }

  _Float16* qo = Qh + ((size_t)b * Nn + n) * Dq;
  _Float16* ko = Kh + ((size_t)b * Nn + n) * Dq;
#pragma unroll
  for (int d = 0; d < Dq; ++d) { qo[d] = (_Float16)qa[d]; ko[d] = (_Float16)ka[d]; }
}

// ---------------------------------------------------------------------------
// Stage 1b: v projection -> f16, stored channel-major [B][C][N]
// ---------------------------------------------------------------------------
__global__ __launch_bounds__(256) void v_proj_kernel(
    const float* __restrict__ ef, const float* __restrict__ sf,
    const float* __restrict__ Vw, const float* __restrict__ Vb,
    _Float16* __restrict__ Vh)
{
  int blk = blockIdx.x;           // Bq * 8 * 16 blocks
  int b   = blk >> 7;
  int cot = (blk >> 4) & 7;
  int nt  = blk & 15;
  int n   = nt * 256 + threadIdx.x;

  float acc[32];
#pragma unroll
  for (int i = 0; i < 32; ++i) acc[i] = Vb[cot * 32 + i];

  for (int c = 0; c < 256; ++c) {
    float xv = ef[((size_t)b * Cc + c) * Nn + n];
#pragma unroll
    for (int i = 0; i < 32; ++i)
      acc[i] = fmaf(Vw[(cot * 32 + i) * 512 + c], xv, acc[i]);
  }
  for (int c = 0; c < 256; ++c) {
    float xv = sf[((size_t)b * Cc + c) * Nn + n];
#pragma unroll
    for (int i = 0; i < 32; ++i)
      acc[i] = fmaf(Vw[(cot * 32 + i) * 512 + 256 + c], xv, acc[i]);
  }
#pragma unroll
  for (int i = 0; i < 32; ++i)
    Vh[((size_t)(b * Cc + cot * 32 + i)) * Nn + n] = (_Float16)acc[i];
}

// ---------------------------------------------------------------------------
// Stage 2: flash attention, 64 keys/iteration, v_wmma_f32_16x16x32_f16,
// software-pipelined K operands (next tile loaded behind the P*V WMMAs).
// Wave = (16 query rows) x (128 output channels). 8 waves/block.
// ---------------------------------------------------------------------------
__global__ __launch_bounds__(256) void attn_kernel(
    const _Float16* __restrict__ Qh, const _Float16* __restrict__ Kh,
    const _Float16* __restrict__ Vh, const float* __restrict__ ef,
    const float* __restrict__ gamma, float* __restrict__ out)
{
  __shared__ __align__(32) char smem_raw[8][2048];   // per-wave scratch

  const int wave = threadIdx.x >> 5;
  const int lane = threadIdx.x & 31;
  const int lr   = lane & 15;
  const int g    = lane >> 4;

  const int gw    = blockIdx.x * 8 + wave;   // 2048 waves total
  const int chalf = gw & 1;                  // which 128-channel half of V
  const int mt    = gw >> 1;                 // 0..1023 query tiles
  const int b     = mt >> 8;
  const int m0    = (mt & 255) << 4;

  // ---- Q A-operand (16x32 f16): two contiguous 16B chunks per lane
  const _Float16* qbase = Qh + ((size_t)(b * Nn + m0 + lr)) * Dq;
  v8h q_lo = *(const v8h*)(qbase + 16 * g);
  v8h q_hi = *(const v8h*)(qbase + 8 + 16 * g);
  v16h a_q;
#pragma unroll
  for (int i = 0; i < 8; ++i) { a_q[i] = q_lo[i]; a_q[i + 8] = q_hi[i]; }

  v8f vzero = {};
  v8f oacc[8];
#pragma unroll
  for (int t = 0; t < 8; ++t) oacc[t] = vzero;

  float rmax[8], rsum[8];
#pragma unroll
  for (int r = 0; r < 8; ++r) { rmax[r] = -__builtin_inff(); rsum[r] = 0.f; }

  _Float16* pbuf = (_Float16*)smem_raw[wave];          // 16m x 64n f16 = 2KB
  const _Float16* kbase = Kh + (size_t)b * Nn * Dq + 16 * g;
  const _Float16* vbase = Vh + ((size_t)(b * Cc + chalf * 128 + lr)) * Nn + 16 * g;

  // ---- prologue: K B-operands for first tile (loop-carried registers)
  v16h kb[4];
  {
    const _Float16* kp = kbase + (size_t)lr * Dq;
#pragma unroll
    for (int j = 0; j < 4; ++j)
      kb[j] = *(const v16h*)(kp + (size_t)(j * 16) * Dq);
  }

  for (int n0 = 0; n0 < Nn; n0 += 64) {
    // ---- S = Qtile x Ktile : four back-to-back WMMAs, operands preloaded
    v8f s[4];
#pragma unroll
    for (int j = 0; j < 4; ++j)
      s[j] = __builtin_amdgcn_wmma_f32_16x16x32_f16(false, a_q, false, kb[j],
                                                    (short)0, vzero, false, false);

    // ---- online softmax (DPP16 row reductions, no LDS traffic)
    float alpha[8];
#pragma unroll
    for (int r = 0; r < 8; ++r) {
      float tmax = fmaxf(fmaxf(s[0][r], s[1][r]), fmaxf(s[2][r], s[3][r]));
      tmax = row_max16(tmax);
      float nm = fmaxf(rmax[r], tmax);
      alpha[r] = __expf(rmax[r] - nm);
      rmax[r]  = nm;
      float rs = 0.f;
#pragma unroll
      for (int j = 0; j < 4; ++j) {
        s[j][r] = __expf(s[j][r] - nm);
        rs += s[j][r];
      }
      rs = row_sum16(rs);
      rsum[r] = rsum[r] * alpha[r] + rs;
    }
#pragma unroll
    for (int t = 0; t < 8; ++t)
#pragma unroll
      for (int r = 0; r < 8; ++r) oacc[t][r] *= alpha[r];

    // ---- issue next K-tile loads now; they complete behind the P*V WMMAs
    if (n0 + 64 < Nn) {
      const _Float16* kp = kbase + (size_t)(n0 + 64 + lr) * Dq;
#pragma unroll
      for (int j = 0; j < 4; ++j)
        kb[j] = *(const v16h*)(kp + (size_t)(j * 16) * Dq);
    }
    if (n0 + 128 < Nn)   // and prefetch the tile after that into cache
      __builtin_prefetch(kbase + (size_t)(n0 + 128 + lr) * Dq, 0, 1);

    // ---- D-layout P -> A-layout P via per-wave LDS tile (16m x 64n f16)
#pragma unroll
    for (int r = 0; r < 8; ++r)
#pragma unroll
      for (int j = 0; j < 4; ++j)
        pbuf[(r + 8 * g) * 64 + j * 16 + lr] = (_Float16)s[j][r];
    __syncthreads();
    const _Float16* pr = pbuf + lr * 64;
    v8h p0l = *(const v8h*)(pr + 16 * g);
    v8h p0h = *(const v8h*)(pr + 8 + 16 * g);
    v8h p1l = *(const v8h*)(pr + 32 + 16 * g);
    v8h p1h = *(const v8h*)(pr + 40 + 16 * g);
    v16h a_p0, a_p1;
#pragma unroll
    for (int i = 0; i < 8; ++i) {
      a_p0[i] = p0l[i]; a_p0[i + 8] = p0h[i];
      a_p1[i] = p1l[i]; a_p1[i + 8] = p1h[i];
    }
    __syncthreads();

    // ---- O += P x V^T : channel tiles in pairs so loads clause together
#pragma unroll
    for (int tp = 0; tp < 4; ++tp) {
      const _Float16* vp0 = vbase + (size_t)((2 * tp) * 16) * Nn + n0;
      const _Float16* vp1 = vbase + (size_t)((2 * tp + 1) * 16) * Nn + n0;
      v16h bv00 = *(const v16h*)(vp0);
      v16h bv01 = *(const v16h*)(vp0 + 32);
      v16h bv10 = *(const v16h*)(vp1);
      v16h bv11 = *(const v16h*)(vp1 + 32);
      oacc[2 * tp] = __builtin_amdgcn_wmma_f32_16x16x32_f16(
          false, a_p0, false, bv00, (short)0, oacc[2 * tp], false, false);
      oacc[2 * tp] = __builtin_amdgcn_wmma_f32_16x16x32_f16(
          false, a_p1, false, bv01, (short)0, oacc[2 * tp], false, false);
      oacc[2 * tp + 1] = __builtin_amdgcn_wmma_f32_16x16x32_f16(
          false, a_p0, false, bv10, (short)0, oacc[2 * tp + 1], false, false);
      oacc[2 * tp + 1] = __builtin_amdgcn_wmma_f32_16x16x32_f16(
          false, a_p1, false, bv11, (short)0, oacc[2 * tp + 1], false, false);
    }
  }

  // ---- epilogue: normalize, transpose via LDS, coalesced gamma*O + ef
  const float g0 = gamma[0];
  float rinv[8];
#pragma unroll
  for (int r = 0; r < 8; ++r) rinv[r] = 1.0f / rsum[r];

  float* obuf = (float*)smem_raw[wave];   // 32c x 16m chunk (2KB)
#pragma unroll 1
  for (int cc = 0; cc < 4; ++cc) {
    __syncthreads();
#pragma unroll
    for (int t2 = 0; t2 < 2; ++t2) {
      int t = cc * 2 + t2;
#pragma unroll
      for (int r = 0; r < 8; ++r)
        obuf[(t2 * 16 + lr) * 16 + r + 8 * g] = oacc[t][r] * rinv[r];
    }
    __syncthreads();
#pragma unroll
    for (int i = 0; i < 16; ++i) {
      int idx = i * 32 + lane;
      int c = chalf * 128 + cc * 32 + (idx >> 4);
      int m = m0 + (idx & 15);
      size_t ga = ((size_t)(b * Cc + c)) * Nn + m;
      out[ga] = g0 * obuf[idx] + ef[ga];
    }
  }
}

// ---------------------------------------------------------------------------
extern "C" void kernel_launch(void* const* d_in, const int* in_sizes, int n_in,
                              void* d_out, int out_size, void* d_ws, size_t ws_size,
                              hipStream_t stream) {
  const float* ef    = (const float*)d_in[0];
  const float* sf    = (const float*)d_in[1];
  const float* At    = (const float*)d_in[2];
  const float* Qw    = (const float*)d_in[3];
  const float* Qb    = (const float*)d_in[4];
  const float* Kw    = (const float*)d_in[5];
  const float* Kb    = (const float*)d_in[6];
  const float* Vw    = (const float*)d_in[7];
  const float* Vb    = (const float*)d_in[8];
  const float* gamma = (const float*)d_in[9];
  float* out = (float*)d_out;

  _Float16* Qh = (_Float16*)d_ws;                 // [B][N][32]  1 MB
  _Float16* Kh = Qh + (size_t)Bq * Nn * Dq;       // [B][N][32]  1 MB
  _Float16* Vh = Kh + (size_t)Bq * Nn * Dq;       // [B][C][N]   8 MB

  qk_proj_kernel<<<Bq * Nn / 256, 256, 0, stream>>>(ef, sf, At, Qw, Qb, Kw, Kb, Qh, Kh);
  v_proj_kernel<<<Bq * 8 * 16, 256, 0, stream>>>(ef, sf, Vw, Vb, Vh);
  attn_kernel<<<Bq * (Nn / 16) * 2 / 8, 256, 0, stream>>>(Qh, Kh, Vh, ef, gamma, out);
}
